// NeuralGraphODELayer_13967233647590
// MI455X (gfx1250) — matrix operand order repaired and checked by reading, not compiled, not run on hardware
//
#include <hip/hip_runtime.h>
#include <math.h>

#define NN 4096
#define HID 64
#define KSPLIT 4
#define KLEN (NN / KSPLIT)       // 1024
#define BROW (KLEN + 8)          // padded LDS row stride (shorts): 2064B = 129*16
#define LDS_BYTES (64 * BROW * 2)  // 132096 B (~129 KB) per workgroup

typedef __bf16 bf16x16 __attribute__((ext_vector_type(16)));
typedef float f32x8 __attribute__((ext_vector_type(8)));
typedef unsigned short u16x8 __attribute__((ext_vector_type(8)));
typedef unsigned short u16x4 __attribute__((ext_vector_type(4)));

union BfPack { bf16x16 v; u16x8 h[2]; };

__device__ __forceinline__ unsigned short f2bf(float f) {
  union { float f; unsigned int u; } x; x.f = f;
  unsigned int u = x.u;
  u += 0x7FFFu + ((u >> 16) & 1u);          // round-to-nearest-even
  return (unsigned short)(u >> 16);
}

// ---------------- S^{1,2} f32 -> bf16 (once per launch) ----------------
__global__ void convert_kernel(const float* __restrict__ Spow,
                               unsigned short* __restrict__ Sbf) {
  size_t base = ((size_t)blockIdx.x * blockDim.x + threadIdx.x) * 4;
  const size_t total = (size_t)2 * NN * NN;
  if (base >= total) return;
  const float4 f = *(const float4*)(Spow + (size_t)NN * NN + base); // skip S^0
  u16x4 o;
  o[0] = f2bf(f.x); o[1] = f2bf(f.y); o[2] = f2bf(f.z); o[3] = f2bf(f.w);
  *(u16x4*)(Sbf + base) = o;
}

// ---------------- Su_m = S^{m+1} @ u  (N x 2, tiny) ----------------
__global__ __launch_bounds__(64) void su_kernel(const float* __restrict__ Spow,
                                                const float* __restrict__ u,
                                                float* __restrict__ Su) {
  __shared__ float red0[64], red1[64];
  const int t = threadIdx.x;
  const int n = blockIdx.x >> 1;
  const int m = blockIdx.x & 1;
  const float* Sr = Spow + (size_t)(m + 1) * NN * NN + (size_t)n * NN;
  float a0 = 0.f, a1 = 0.f;
  for (int j = t; j < NN; j += 64) {
    float s = Sr[j];
    a0 += s * u[2 * j];
    a1 += s * u[2 * j + 1];
  }
  red0[t] = a0; red1[t] = a1;
  __syncthreads();
  for (int off = 32; off > 0; off >>= 1) {
    if (t < off) { red0[t] += red0[t + off]; red1[t] += red1[t + off]; }
    __syncthreads();
  }
  if (t == 0) {
    Su[(size_t)m * NN * 2 + 2 * n]     = red0[0];
    Su[(size_t)m * NN * 2 + 2 * n + 1] = red1[0];
  }
}

// ---------------- y0 = x + graph_filter(u, W_in) + b_in ----------------
__global__ void y0_kernel(const float* __restrict__ x, const float* __restrict__ u,
                          const float* __restrict__ Su, const float* __restrict__ Win,
                          const float* __restrict__ bin, float* __restrict__ y) {
  int i = blockIdx.x * blockDim.x + threadIdx.x;
  if (i >= NN * HID) return;
  int n = i >> 6, e = i & 63;
  float o = x[i] + bin[e];
  o += u[2 * n]           * Win[0 * 128 + e] + u[2 * n + 1]           * Win[0 * 128 + 64 + e];
  o += Su[2 * n]          * Win[1 * 128 + e] + Su[2 * n + 1]          * Win[1 * 128 + 64 + e];
  o += Su[NN * 2 + 2 * n] * Win[2 * 128 + e] + Su[NN * 2 + 2 * n + 1] * Win[2 * 128 + 64 + e];
  y[i] = o;
}

// ------- stage combine: ys = y + sum(c_j * k_j); also write ysT (bf16) -------
__global__ __launch_bounds__(256) void combine_kernel(
    const float* __restrict__ y, float* __restrict__ yOut, int writeY,
    const float* __restrict__ k1, const float* __restrict__ k2,
    const float* __restrict__ k3, const float* __restrict__ k4,
    const float* __restrict__ k5, const float* __restrict__ k6,
    float c1, float c2, float c3, float c4, float c5, float c6,
    float* __restrict__ ys, unsigned short* __restrict__ ysT) {
  __shared__ unsigned short tile[64][72];
  const int tx = threadIdx.x;   // 0..63 (HID column)
  const int ty = threadIdx.y;   // 0..3
  const int rb = blockIdx.x * 64;
  for (int r = ty; r < 64; r += 4) {
    size_t idx = (size_t)(rb + r) * HID + tx;
    float v = y[idx];
    if (c1 != 0.f) v += c1 * k1[idx];
    if (c2 != 0.f) v += c2 * k2[idx];
    if (c3 != 0.f) v += c3 * k3[idx];
    if (c4 != 0.f) v += c4 * k4[idx];
    if (c5 != 0.f) v += c5 * k5[idx];
    if (c6 != 0.f) v += c6 * k6[idx];
    ys[idx] = v;
    if (writeY) yOut[idx] = v;
    tile[tx][r] = f2bf(v);
  }
  __syncthreads();
  for (int c = ty; c < 64; c += 4)
    ysT[(size_t)c * NN + rb + tx] = tile[c][tx];
}

// ------------- WMMA: Pp[ks] = S^1[:,kslice] @ ys[kslice,:], Qp likewise -------------
// Grid: 64 row-groups x KSPLIT. Block = 8 waves; wave w: m = w>>2, rowSub = w&3.
// B slice (64 x KLEN bf16, shared by all 8 waves) staged in LDS once;
// inner loop: 2 global A loads (double-buffered) + 8 ds_load_b128 + 4 WMMA per chunk.
__global__ __launch_bounds__(256) void gemm_kernel(const unsigned short* __restrict__ Sbf,
                                                   const unsigned short* __restrict__ ysT,
                                                   float* __restrict__ Pp,
                                                   float* __restrict__ Qp) {
  extern __shared__ unsigned short bs[];           // [64][BROW]
  const int tid  = threadIdx.x;
  const int wave = tid >> 5;
  const int lane = tid & 31;
  const int ks = blockIdx.x & (KSPLIT - 1);
  const int rowGroup = blockIdx.x / KSPLIT;        // 0..63
  const int m = wave >> 2;
  const int rowBase = (rowGroup * 4 + (wave & 3)) * 16;
  const unsigned short* Sm = Sbf + (size_t)m * NN * NN;
  float* out = (m == 0 ? Pp : Qp) + (size_t)ks * NN * HID;

  // ---- cooperative fill of B slice into LDS ----
  for (int i = tid; i < 64 * (KLEN / 8); i += 256) {      // 8192 chunks of 8 shorts
    int row = i >> 7, chunk = i & 127;                    // KLEN/8 = 128
    const u16x8 v = *(const u16x8*)(ysT + (size_t)row * NN + (size_t)ks * KLEN + chunk * 8);
    *(u16x8*)(bs + row * BROW + chunk * 8) = v;
  }
  __syncthreads();

  const int half = lane >> 4;            // 0: lanes 0-15, 1: lanes 16-31
  const int aK   = half * 8;             // A: K base 0 or 8 (VGPR 0-3); +16 for VGPR 4-7
  const int bK   = half * 8;             // B: K base 0 or 8 (elems 0-7); +16 for elems 8-15
  const int bCol = lane & 15;
  const int aRow = rowBase + (lane & 15);

  const unsigned short* aP = Sm + (size_t)aRow * NN + (size_t)ks * KLEN + aK;
  const unsigned short* bL[4];
#pragma unroll
  for (int ct = 0; ct < 4; ++ct)
    bL[ct] = bs + (ct * 16 + bCol) * BROW + bK;

  f32x8 acc[4] = {{}, {}, {}, {}};
  BfPack a0, a1, b0[4], b1[4];

  // prologue: chunk 0
  a0.h[0] = *(const u16x8*)(aP);
  a0.h[1] = *(const u16x8*)(aP + 16);
#pragma unroll
  for (int ct = 0; ct < 4; ++ct) {
    b0[ct].h[0] = *(const u16x8*)(bL[ct]);
    b0[ct].h[1] = *(const u16x8*)(bL[ct] + 16);
  }

  for (int kc = 0; kc < KLEN; kc += 64) {
    __builtin_prefetch(aP + kc + 512, 0, 1);      // stream S ahead (HBM first pass)
    // chunk kc+32 -> buffer 1
    a1.h[0] = *(const u16x8*)(aP + kc + 32);
    a1.h[1] = *(const u16x8*)(aP + kc + 48);
#pragma unroll
    for (int ct = 0; ct < 4; ++ct) {
      b1[ct].h[0] = *(const u16x8*)(bL[ct] + kc + 32);
      b1[ct].h[1] = *(const u16x8*)(bL[ct] + kc + 48);
    }
#pragma unroll
    for (int ct = 0; ct < 4; ++ct)
      acc[ct] = __builtin_amdgcn_wmma_f32_16x16x32_bf16(
          false, a0.v, false, b0[ct].v, (short)0, acc[ct], false, false);
    // chunk kc+64 -> buffer 0
    if (kc + 64 < KLEN) {
      a0.h[0] = *(const u16x8*)(aP + kc + 64);
      a0.h[1] = *(const u16x8*)(aP + kc + 80);
#pragma unroll
      for (int ct = 0; ct < 4; ++ct) {
        b0[ct].h[0] = *(const u16x8*)(bL[ct] + kc + 64);
        b0[ct].h[1] = *(const u16x8*)(bL[ct] + kc + 80);
      }
    }
#pragma unroll
    for (int ct = 0; ct < 4; ++ct)
      acc[ct] = __builtin_amdgcn_wmma_f32_16x16x32_bf16(
          false, a1.v, false, b1[ct].v, (short)0, acc[ct], false, false);
  }

#pragma unroll
  for (int ct = 0; ct < 4; ++ct)
#pragma unroll
    for (int v = 0; v < 8; ++v) {
      int r = rowBase + v + half * 8;
      out[(size_t)r * HID + ct * 16 + bCol] = acc[ct][v];
    }
}

// ------------- k = f(ys) = -decay*ys + GF(ys) + MLP(ys); sums K-split partials -------------
__global__ __launch_bounds__(256) void f_kernel(
    const float* __restrict__ ys, const float* __restrict__ Pp, const float* __restrict__ Qp,
    const float* __restrict__ Wg, const float* __restrict__ bg,
    const float* __restrict__ W1, const float* __restrict__ b1,
    const float* __restrict__ W2, const float* __restrict__ b2,
    const float* __restrict__ decay, float* __restrict__ kout) {
  __shared__ float sy[32][64], sp[32][64], sq[32][64], st[32][64];
  const int tid = threadIdx.x;
  const int rowBase = blockIdx.x * 32;
  const size_t stride = (size_t)NN * HID;
  for (int i = tid; i < 32 * 64; i += 256) {
    int r = i >> 6, c = i & 63;
    size_t g = (size_t)(rowBase + r) * HID + c;
    sy[r][c] = ys[g];
    float p = Pp[g], q = Qp[g];
#pragma unroll
    for (int s = 1; s < KSPLIT; ++s) { p += Pp[s * stride + g]; q += Qp[s * stride + g]; }
    sp[r][c] = p; sq[r][c] = q;
  }
  __syncthreads();
  const int e = tid & 63;
  const int rr = tid >> 6;
  for (int rb = 0; rb < 32; rb += 4) {
    int r = rb + rr;
    float a = b1[e];
    for (int d = 0; d < 64; ++d) a += sy[r][d] * W1[d * 64 + e];
    st[r][e] = tanhf(a);
  }
  __syncthreads();
  for (int rb = 0; rb < 32; rb += 4) {
    int r = rb + rr;
    float o = bg[e] + b2[e] - decay[e] * sy[r][e];
    for (int d = 0; d < 64; ++d) {
      o += sy[r][d] * Wg[d * 64 + e];
      o += sp[r][d] * Wg[4096 + d * 64 + e];
      o += sq[r][d] * Wg[8192 + d * 64 + e];
      o += st[r][d] * W2[d * 64 + e];
    }
    kout[(size_t)(rowBase + r) * HID + e] = o;
  }
}

__global__ void clip_kernel(const float* __restrict__ y, float* __restrict__ out) {
  int i = blockIdx.x * blockDim.x + threadIdx.x;
  if (i < NN * HID) out[i] = fminf(1.f, fmaxf(-1.f, y[i]));
}

extern "C" void kernel_launch(void* const* d_in, const int* in_sizes, int n_in,
                              void* d_out, int out_size, void* d_ws, size_t ws_size,
                              hipStream_t stream) {
  (void)in_sizes; (void)n_in; (void)out_size; (void)ws_size;
  const float* x     = (const float*)d_in[0];
  const float* u     = (const float*)d_in[1];
  const float* Spow  = (const float*)d_in[2];
  const float* Win   = (const float*)d_in[3];
  const float* bin   = (const float*)d_in[4];
  const float* Wg    = (const float*)d_in[5];
  const float* bg    = (const float*)d_in[6];
  const float* W1    = (const float*)d_in[7];
  const float* b1    = (const float*)d_in[8];
  const float* W2    = (const float*)d_in[9];
  const float* b2    = (const float*)d_in[10];
  const float* decay = (const float*)d_in[11];

  char* ws = (char*)d_ws;
  size_t off = 0;
  auto alloc = [&](size_t bytes) -> char* {
    char* p = ws + off;
    off += (bytes + 255) & ~(size_t)255;
    return p;
  };
  unsigned short* Sbf = (unsigned short*)alloc((size_t)2 * NN * NN * 2);
  unsigned short* ysT = (unsigned short*)alloc((size_t)HID * NN * 2);
  float* ys = (float*)alloc((size_t)NN * HID * 4);
  float* y  = (float*)alloc((size_t)NN * HID * 4);
  float* Pp = (float*)alloc((size_t)KSPLIT * NN * HID * 4);
  float* Qp = (float*)alloc((size_t)KSPLIT * NN * HID * 4);
  float* Su = (float*)alloc((size_t)2 * NN * 2 * 4);
  float* kb[6];
  for (int j = 0; j < 6; ++j) kb[j] = (float*)alloc((size_t)NN * HID * 4);

  const double h = 0.05 / 4.0;  // DT*N_STEPS/N_INT
  // C[0]: stage-1 passthrough; C[1..5]: dopri5 A rows (x h); C[6]: B row (x h)
  const double Cd[7][6] = {
      {0, 0, 0, 0, 0, 0},
      {h * (1.0 / 5.0), 0, 0, 0, 0, 0},
      {h * (3.0 / 40.0), h * (9.0 / 40.0), 0, 0, 0, 0},
      {h * (44.0 / 45.0), h * (-56.0 / 15.0), h * (32.0 / 9.0), 0, 0, 0},
      {h * (19372.0 / 6561.0), h * (-25360.0 / 2187.0), h * (64448.0 / 6561.0),
       h * (-212.0 / 729.0), 0, 0},
      {h * (9017.0 / 3168.0), h * (-355.0 / 33.0), h * (46732.0 / 5247.0),
       h * (49.0 / 176.0), h * (-5103.0 / 18656.0), 0},
      {h * (35.0 / 384.0), 0, h * (500.0 / 1113.0), h * (125.0 / 192.0),
       h * (-2187.0 / 6784.0), h * (11.0 / 84.0)}};
  float C[7][6];
  for (int i = 0; i < 7; ++i)
    for (int j = 0; j < 6; ++j) C[i][j] = (float)Cd[i][j];

  // ---- init ----
  {
    size_t nconv = ((size_t)2 * NN * NN) / 4;
    convert_kernel<<<(unsigned)((nconv + 255) / 256), 256, 0, stream>>>(Spow, Sbf);
  }
  su_kernel<<<NN * 2, 64, 0, stream>>>(Spow, u, Su);
  y0_kernel<<<(NN * HID + 255) / 256, 256, 0, stream>>>(x, u, Su, Win, bin, y);
  combine_kernel<<<64, dim3(64, 4), 0, stream>>>(
      y, y, 0, kb[0], kb[1], kb[2], kb[3], kb[4], kb[5],
      C[0][0], C[0][1], C[0][2], C[0][3], C[0][4], C[0][5], ys, ysT);

  // ---- 4 dopri5 sub-steps ----
  for (int step = 0; step < 4; ++step) {
    for (int s = 1; s <= 6; ++s) {
      if (s > 1) {
        combine_kernel<<<64, dim3(64, 4), 0, stream>>>(
            y, y, 0, kb[0], kb[1], kb[2], kb[3], kb[4], kb[5],
            C[s - 1][0], C[s - 1][1], C[s - 1][2], C[s - 1][3], C[s - 1][4],
            C[s - 1][5], ys, ysT);
      }
      gemm_kernel<<<64 * KSPLIT, 256, LDS_BYTES, stream>>>(Sbf, ysT, Pp, Qp);
      f_kernel<<<128, 256, 0, stream>>>(ys, Pp, Qp, Wg, bg, W1, b1, W2, b2,
                                        decay, kb[s - 1]);
    }
    // y <- y + h*sum(B_j k_j); also pre-stage ys/ysT for next step's k1
    combine_kernel<<<64, dim3(64, 4), 0, stream>>>(
        y, y, 1, kb[0], kb[1], kb[2], kb[3], kb[4], kb[5],
        C[6][0], C[6][1], C[6][2], C[6][3], C[6][4], C[6][5], ys, ysT);
  }

  clip_kernel<<<(NN * HID + 255) / 256, 256, 0, stream>>>(y, (float*)d_out);
}
